// PQLinearShared_33938831573579
// MI455X (gfx1250) — matrix operand order repaired and checked by reading, not compiled, not run on hardware
//
#include <hip/hip_runtime.h>

// Problem sizes (fixed by the reference)
#define O_DIM 2048   // output features (GEMM N)
#define I_DIM 4096   // input features  (GEMM K)
#define D_DIM 8      // PQ group dim
#define K_CB  256    // codebook entries
#define M_DIM 8192   // B*S tokens      (GEMM M)

// GEMM tiling
#define BM 128
#define BN 128
#define BK 32        // one wmma 16x16x32 K-step
#define LDSP 40      // padded LDS row stride (elements): 80B = 5*16B, bank-conflict-free

typedef float  v8f   __attribute__((ext_vector_type(8)));
typedef __bf16 v8bf  __attribute__((ext_vector_type(8)));
typedef __bf16 v16bf __attribute__((ext_vector_type(16)));

union BF16Frag { v16bf v; v8bf h[2]; };

__device__ __forceinline__ unsigned short f32_to_bf16_rne(float f) {
  unsigned int u = __float_as_uint(f);
  u += 0x7FFFu + ((u >> 16) & 1u);
  return (unsigned short)(u >> 16);
}

// ---------------------------------------------------------------------------
// Kernel 1: PQ quantization.  One 8-dim group per lane; codebook + codeword
// norms staged in LDS; argmin matches the reference (first-min on ties,
// d2 = (||g||^2 - 2 g.c) + ||c||^2).  Output: dequantized wq in bf16 (ws).
// ---------------------------------------------------------------------------
__global__ __launch_bounds__(256) void pq_quant_kernel(
    const float* __restrict__ weight, const float* __restrict__ codebook,
    const float* __restrict__ row_scale, unsigned short* __restrict__ wq_bf16)
{
  __shared__ float cb[K_CB * D_DIM];
  __shared__ float cnorm[K_CB];
  const int t = threadIdx.x;
  for (int i = t; i < K_CB * D_DIM; i += 256) cb[i] = codebook[i];
  __syncthreads();
  {
    float s = 0.f;
    #pragma unroll
    for (int k = 0; k < D_DIM; ++k) { float c = cb[t * D_DIM + k]; s = fmaf(c, c, s); }
    cnorm[t] = s;   // blockDim == K_CB == 256
  }
  __syncthreads();

  const int g = blockIdx.x * 256 + t;     // group id in [0, O*I/D)
  const int o = g >> 9;                   // I/D = 512 groups per row
  const int j = g & 511;
  const float* wp = weight + (size_t)o * I_DIM + j * D_DIM;
  const float rs = row_scale[o];

  float gv[D_DIM];
  float gg = 0.f;
  #pragma unroll
  for (int k = 0; k < D_DIM; ++k) {
    float v = wp[k] / rs;                 // true division, as in reference
    gv[k] = v;
    gg = fmaf(v, v, gg);
  }

  float best = 3.4e38f;
  int bidx = 0;
  for (int c = 0; c < K_CB; ++c) {
    float dot = 0.f;
    #pragma unroll
    for (int k = 0; k < D_DIM; ++k) dot = fmaf(gv[k], cb[c * D_DIM + k], dot);
    float d2 = (gg - 2.f * dot) + cnorm[c];
    if (d2 < best) { best = d2; bidx = c; }   // strict '<' => first min wins
  }

  unsigned short* dst = wq_bf16 + (size_t)o * I_DIM + j * D_DIM;
  #pragma unroll
  for (int k = 0; k < D_DIM; ++k)
    dst[k] = f32_to_bf16_rne(cb[bidx * D_DIM + k] * rs);
}

// ---------------------------------------------------------------------------
// Kernel 2: out[M, O] = x[M, I] @ wq[O, I]^T + bias, bf16 WMMA / f32 accum.
// 256 threads = 8 waves; block tile 128x128; wave tile 32x64 (2x4 wmma tiles);
// double-buffered LDS with register staging, one barrier per K-iteration.
// ---------------------------------------------------------------------------
__global__ __launch_bounds__(256) void pq_gemm_kernel(
    const float* __restrict__ x, const unsigned short* __restrict__ wq_bf16,
    const float* __restrict__ bias, float* __restrict__ out)
{
  __shared__ unsigned short As[2][BM * LDSP];
  __shared__ unsigned short Bs[2][BN * LDSP];

  const int n0 = blockIdx.x * BN;
  const int m0 = blockIdx.y * BM;
  const int t    = threadIdx.x;
  const int lane = t & 31;
  const int wave = t >> 5;
  const int wm = wave & 3;      // 4 waves along M, 32 rows each
  const int wn = wave >> 2;     // 2 waves along N, 64 cols each
  const int h  = lane >> 4;     // half-wave select
  const int lr = lane & 15;

  v8f acc[2][4];
  #pragma unroll
  for (int i = 0; i < 2; ++i)
    #pragma unroll
    for (int j = 0; j < 4; ++j)
      #pragma unroll
      for (int r = 0; r < 8; ++r) acc[i][j][r] = 0.f;

  float4  areg[4];
  ushort4 breg[4];

  // Cooperative tile load mapping: 128 rows x 8 chunks-of-4, 4 chunks/thread.
  int lrow[4], lq[4];
  #pragma unroll
  for (int i = 0; i < 4; ++i) {
    int linear = t + 256 * i;
    lrow[i] = linear >> 3;
    lq[i]   = linear & 7;
  }

  // ---- prologue: fetch k0 = 0 into buffer 0
  #pragma unroll
  for (int i = 0; i < 4; ++i) {
    areg[i] = *(const float4*)(x + (size_t)(m0 + lrow[i]) * I_DIM + lq[i] * 4);
    breg[i] = *(const ushort4*)(wq_bf16 + (size_t)(n0 + lrow[i]) * I_DIM + lq[i] * 4);
  }
  #pragma unroll
  for (int i = 0; i < 4; ++i) {
    ushort4 av;
    av.x = f32_to_bf16_rne(areg[i].x);
    av.y = f32_to_bf16_rne(areg[i].y);
    av.z = f32_to_bf16_rne(areg[i].z);
    av.w = f32_to_bf16_rne(areg[i].w);
    *(ushort4*)&As[0][lrow[i] * LDSP + lq[i] * 4] = av;
    *(ushort4*)&Bs[0][lrow[i] * LDSP + lq[i] * 4] = breg[i];
  }
  __syncthreads();

  for (int k0 = 0; k0 < I_DIM; k0 += BK) {
    const int buf = (k0 / BK) & 1;
    const bool has_next = (k0 + BK) < I_DIM;

    // Issue next-tile global loads early (latency overlap with WMMA).
    if (has_next) {
      #pragma unroll
      for (int i = 0; i < 4; ++i) {
        areg[i] = *(const float4*)(x + (size_t)(m0 + lrow[i]) * I_DIM + (k0 + BK) + lq[i] * 4);
        breg[i] = *(const ushort4*)(wq_bf16 + (size_t)(n0 + lrow[i]) * I_DIM + (k0 + BK) + lq[i] * 4);
      }
    }

    // Fragment loads per documented wave32 layouts.
    // A (16x32, MxK): lane(h, m=lr) holds K in [8h, 8h+8) and [16+8h, 16+8h+8).
    BF16Frag afrag[2];
    #pragma unroll
    for (int mt = 0; mt < 2; ++mt) {
      const unsigned short* base = &As[buf][(wm * 32 + mt * 16 + lr) * LDSP + 8 * h];
      afrag[mt].h[0] = *(const v8bf*)(base);        // K = 8h .. 8h+7
      afrag[mt].h[1] = *(const v8bf*)(base + 16);   // K = 16+8h .. 16+8h+7
    }
    // B (32x16, KxN): lane(h, n=lr) holds K in [8h, 8h+8) and [16+8h, 16+8h+8).
    BF16Frag bfrag[4];
    #pragma unroll
    for (int nt = 0; nt < 4; ++nt) {
      const unsigned short* base = &Bs[buf][(wn * 64 + nt * 16 + lr) * LDSP + 8 * h];
      bfrag[nt].h[0] = *(const v8bf*)(base);        // K = 8h .. 8h+7
      bfrag[nt].h[1] = *(const v8bf*)(base + 16);   // K = 16+8h .. 16+8h+7
    }

    #pragma unroll
    for (int mt = 0; mt < 2; ++mt)
      #pragma unroll
      for (int nt = 0; nt < 4; ++nt)
        acc[mt][nt] = __builtin_amdgcn_wmma_f32_16x16x32_bf16(
            false, afrag[mt].v, false, bfrag[nt].v,
            (short)0, acc[mt][nt], false, false);

    // Stage next tile into the other buffer (no conflict with in-flight reads).
    if (has_next) {
      #pragma unroll
      for (int i = 0; i < 4; ++i) {
        ushort4 av;
        av.x = f32_to_bf16_rne(areg[i].x);
        av.y = f32_to_bf16_rne(areg[i].y);
        av.z = f32_to_bf16_rne(areg[i].z);
        av.w = f32_to_bf16_rne(areg[i].w);
        *(ushort4*)&As[buf ^ 1][lrow[i] * LDSP + lq[i] * 4] = av;
        *(ushort4*)&Bs[buf ^ 1][lrow[i] * LDSP + lq[i] * 4] = breg[i];
      }
    }
    __syncthreads();
  }

  // Epilogue: C/D layout => VGPR r holds (M = r + 8h, N = lr) of each tile.
  #pragma unroll
  for (int nt = 0; nt < 4; ++nt) {
    const int col = n0 + wn * 64 + nt * 16 + lr;
    const float bv = bias[col];
    #pragma unroll
    for (int mt = 0; mt < 2; ++mt) {
      const int rbase = m0 + wm * 32 + mt * 16 + 8 * h;
      #pragma unroll
      for (int r = 0; r < 8; ++r)
        out[(size_t)(rbase + r) * O_DIM + col] = acc[mt][nt][r] + bv;
    }
  }
}

// ---------------------------------------------------------------------------
extern "C" void kernel_launch(void* const* d_in, const int* in_sizes, int n_in,
                              void* d_out, int out_size, void* d_ws, size_t ws_size,
                              hipStream_t stream) {
  (void)in_sizes; (void)n_in; (void)out_size; (void)ws_size;
  const float* x         = (const float*)d_in[0];
  const float* weight    = (const float*)d_in[1];
  const float* codebook  = (const float*)d_in[2];
  const float* row_scale = (const float*)d_in[3];
  const float* bias      = (const float*)d_in[4];
  float* out             = (float*)d_out;
  unsigned short* wq     = (unsigned short*)d_ws;   // O*I bf16 = 16 MB scratch

  // 1) Quantize weight -> bf16 wq in workspace.
  pq_quant_kernel<<<dim3((O_DIM * I_DIM / D_DIM) / 256), 256, 0, stream>>>(
      weight, codebook, row_scale, wq);

  // 2) bf16 WMMA GEMM with f32 accumulate + bias.
  pq_gemm_kernel<<<dim3(O_DIM / BN, M_DIM / BM), 256, 0, stream>>>(
      x, wq, bias, out);
}
